// GNNEncoder_51737176047987
// MI455X (gfx1250) — matrix operand order repaired and checked by reading, not compiled, not run on hardware
//
#include <hip/hip_runtime.h>

// ---------------------------------------------------------------------------
// GNN encoder for MI455X (gfx1250, wave32, WMMA).
// Per-edge 2-layer MLP via v_wmma_f32_16x16x32_f16 on 16-edge tiles.
//  - weights staged per block in LDS column-major f16: B fragment = 1 half16 load
//  - inter-layer node features kept in f16 globally: gather = 4 b128 loads/lane
//  - edge_attr pre-converted to f16 once
//  - f32 accumulate, atomic f32 scatter, mean+ReLU node update
// ---------------------------------------------------------------------------

typedef _Float16 half16 __attribute__((ext_vector_type(16)));
typedef _Float16 half8  __attribute__((ext_vector_type(8)));
typedef float    float8 __attribute__((ext_vector_type(8)));

#define GNN_N_NODES 100000
#define GNN_N_EDGES 1600000
#define WPB 4               // waves per block (128 threads)
#define TILES_PER_WAVE 8    // grid-stride amortization of LDS weight staging

// --------------------------- helper kernels --------------------------------

__global__ void gnn_zero_kernel(float* __restrict__ p, int total) {
    int i = blockIdx.x * blockDim.x + threadIdx.x;
    if (i < total) p[i] = 0.0f;
}

__global__ void gnn_cvt_f16_kernel(const float* __restrict__ src,
                                   _Float16* __restrict__ dst, int total) {
    int i = blockIdx.x * blockDim.x + threadIdx.x;
    if (i < total) dst[i] = (_Float16)src[i];
}

__global__ void gnn_degree_kernel(const long long* __restrict__ edge_index,
                                  float* __restrict__ cnt, int nEdges) {
    int e = blockIdx.x * blockDim.x + threadIdx.x;
    if (e < nEdges) {
        int d = (int)edge_index[(size_t)nEdges + e];   // dst row
        atomicAdd(&cnt[d], 1.0f);
    }
}

// mean + relu -> f16 node features, and re-zero accum for the next layer
template <int D>
__global__ void gnn_node_finish_kernel(const float* __restrict__ cnt,
                                       float* __restrict__ accum,
                                       _Float16* __restrict__ feat, int total) {
    int i = blockIdx.x * blockDim.x + threadIdx.x;
    if (i < total) {
        int node = i / D;
        float c = fmaxf(cnt[node], 1.0f);
        float v = fmaxf(accum[i] / c, 0.0f);
        feat[i]  = (_Float16)v;
        accum[i] = 0.0f;
    }
}

template <int D>
__global__ void gnn_final_kernel(const float* __restrict__ cnt,
                                 const float* __restrict__ accum,
                                 float* __restrict__ out, int total) {
    int i = blockIdx.x * blockDim.x + threadIdx.x;
    if (i < total) {
        int node = i / D;
        out[i] = accum[i] / fmaxf(cnt[node], 1.0f);
    }
}

// --------------------------- edge MLP kernel -------------------------------
// One wave32 handles 16-edge tiles (grid-stride):
//   feat[16 x KIN] -> relu(@Wa + ba)[16 x HIDD] -> (@Wb + bb)[16 x OUTD]
//   then atomicAdd scatter into accum[dst][OUTD].
// FEAT==64 expects nfeat as f16 [node][64]; FEAT==6 expects f32 (layer 1 x).

template <int FEAT, int HIDD, int OUTD>
__global__ __launch_bounds__(32 * WPB)
void gnn_edge_mlp_kernel(const void* __restrict__ nfeat,
                         const long long* __restrict__ edge_index,
                         const _Float16* __restrict__ ea16,
                         const float* __restrict__ Wa, const float* __restrict__ ba,
                         const float* __restrict__ Wb, const float* __restrict__ bb,
                         float* __restrict__ accum, int nEdges, int nTiles) {
    constexpr int KIN  = FEAT + 3;
    constexpr int KC1  = (KIN + 31) / 32;   // k-chunks GEMM1
    constexpr int KPAD = KC1 * 32;
    constexpr int NT1  = HIDD / 16;         // n-tiles GEMM1
    constexpr int KC2  = HIDD / 32;         // k-chunks GEMM2
    constexpr int NT2  = OUTD / 16;         // n-tiles GEMM2

    // weights column-major in LDS: one half16 load per WMMA B fragment
    __shared__ __align__(32) _Float16 s_wa[HIDD][KPAD];
    __shared__ __align__(32) _Float16 s_wb[OUTD][HIDD];
    __shared__ float s_ba[HIDD];
    __shared__ float s_bb[OUTD];
    // per-wave tiles
    __shared__ __align__(32) _Float16 s_feat[WPB][16][KPAD];
    __shared__ __align__(32) _Float16 s_hid [WPB][16][HIDD];
    __shared__ int s_src[WPB][16];
    __shared__ int s_dst[WPB][16];

    const int tid    = threadIdx.x;
    const int wave   = tid >> 5;
    const int lane   = tid & 31;
    const int lanelo = lane & 15;
    const int hi     = lane >> 4;

    // ---- stage weights + biases once per block --------------------------
    for (int i = tid; i < HIDD * KPAD; i += 32 * WPB) {
        int n = i / KPAD, k = i % KPAD;
        s_wa[n][k] = (_Float16)((k < KIN) ? Wa[k * HIDD + n] : 0.0f);
    }
    for (int i = tid; i < OUTD * HIDD; i += 32 * WPB) {
        int n = i / HIDD, k = i % HIDD;
        s_wb[n][k] = (_Float16)Wb[k * OUTD + n];
    }
    for (int i = tid; i < HIDD; i += 32 * WPB) s_ba[i] = ba[i];
    for (int i = tid; i < OUTD; i += 32 * WPB) s_bb[i] = bb[i];

    // ---- zero this wave's s_feat once (padding is invariant zero) -------
    {
        _Float16* flat = &s_feat[wave][0][0];
        half8 z = {};
        for (int i = lane; i < (16 * KPAD) / 8; i += 32)
            *(half8*)&flat[i * 8] = z;
    }
    __syncthreads();

    const int stride = (int)gridDim.x * WPB;
    const int iters  = (nTiles + stride - 1) / stride;   // uniform -> barriers safe
    int tile = (int)blockIdx.x * WPB + wave;

    for (int it = 0; it < iters; ++it, tile += stride) {
        const int e0 = tile * 16;

        // edge endpoints for this tile (guarded; phantom tiles -> src 0 / dst -1)
        if (lane < 16) {
            int e   = e0 + lane;
            bool ok = (tile < nTiles) && (e < nEdges);
            s_src[wave][lane] = ok ? (int)edge_index[e] : 0;
            s_dst[wave][lane] = ok ? (int)edge_index[(size_t)nEdges + e] : -1;
        }
        __syncthreads();

        // ---- gather [feat[src] | edge_attr] -> s_feat (k >= KIN stays 0) ----
        if constexpr (FEAT == 64) {
            const _Float16* nf = (const _Float16*)nfeat;
            const int sub = lane & 7;      // 8 lanes cover one 64-f16 row
            const int r4  = lane >> 3;     // 4 rows per pass
#pragma unroll
            for (int i = 0; i < 4; ++i) {
                int row = i * 4 + r4;
                const _Float16* p = nf + (size_t)s_src[wave][row] * 64;
                half8 v = *(const half8*)&p[sub * 8];
                *(half8*)&s_feat[wave][row][sub * 8] = v;
            }
            // k in [64,67): 3 f16 edge attrs (48 values, 2 iters/lane)
#pragma unroll
            for (int idx = lane; idx < 48; idx += 32) {
                int row = idx / 3, k = idx - row * 3;
                int e = e0 + row;
                _Float16 v = (_Float16)0.0f;
                if (tile < nTiles && e < nEdges) v = ea16[(size_t)e * 3 + k];
                s_feat[wave][row][64 + k] = v;
            }
        } else {
            const float* nf = (const float*)nfeat;
            for (int idx = lane; idx < 16 * KIN; idx += 32) {
                int row = idx / KIN, k = idx - row * KIN;
                int e = e0 + row;
                float v = 0.0f;
                if (tile < nTiles && e < nEdges) {
                    if (k < FEAT) v = nf[(size_t)s_src[wave][row] * FEAT + k];
                    else          v = (float)ea16[(size_t)e * 3 + (k - FEAT)];
                }
                s_feat[wave][row][k] = (_Float16)v;
            }
        }
        __syncthreads();

        // ---- A fragments GEMM1: two aligned half8 runs per k-chunk ------
        half16 a1[KC1];
#pragma unroll
        for (int c = 0; c < KC1; ++c) {
            half8 lo = *(const half8*)&s_feat[wave][lanelo][c * 32 + hi * 16];
            half8 hh = *(const half8*)&s_feat[wave][lanelo][c * 32 + hi * 16 + 8];
            a1[c] = __builtin_shufflevector(lo, hh, 0, 1, 2, 3, 4, 5, 6, 7,
                                                    8, 9, 10, 11, 12, 13, 14, 15);
        }

        // ---- GEMM1 + bias + relu -> s_hid (f16, D layout) ---------------
#pragma unroll
        for (int nt = 0; nt < NT1; ++nt) {
            float8 acc = {};
            const int n = nt * 16 + lanelo;
#pragma unroll
            for (int c = 0; c < KC1; ++c) {
                half16 b = *(const half16*)&s_wa[n][c * 32 + hi * 16];
                acc = __builtin_amdgcn_wmma_f32_16x16x32_f16(
                    false, a1[c], false, b, (short)0, acc, false, false);
            }
            float bias = s_ba[n];
#pragma unroll
            for (int r = 0; r < 8; ++r)
                s_hid[wave][r + hi * 8][n] = (_Float16)fmaxf(acc[r] + bias, 0.0f);
        }
        __syncthreads();

        // ---- A fragments GEMM2 from s_hid -------------------------------
        half16 a2[KC2];
#pragma unroll
        for (int c = 0; c < KC2; ++c) {
            half8 lo = *(const half8*)&s_hid[wave][lanelo][c * 32 + hi * 16];
            half8 hh = *(const half8*)&s_hid[wave][lanelo][c * 32 + hi * 16 + 8];
            a2[c] = __builtin_shufflevector(lo, hh, 0, 1, 2, 3, 4, 5, 6, 7,
                                                    8, 9, 10, 11, 12, 13, 14, 15);
        }

        int dstv[8];
#pragma unroll
        for (int r = 0; r < 8; ++r) dstv[r] = s_dst[wave][r + hi * 8];

        // ---- GEMM2 + bias -> atomic scatter -----------------------------
#pragma unroll
        for (int ot = 0; ot < NT2; ++ot) {
            float8 acc = {};
            const int n = ot * 16 + lanelo;
#pragma unroll
            for (int c = 0; c < KC2; ++c) {
                half16 b = *(const half16*)&s_wb[n][c * 32 + hi * 16];
                acc = __builtin_amdgcn_wmma_f32_16x16x32_f16(
                    false, a2[c], false, b, (short)0, acc, false, false);
            }
            float bias = s_bb[n];
#pragma unroll
            for (int r = 0; r < 8; ++r) {
                if (dstv[r] >= 0)
                    atomicAdd(&accum[(size_t)dstv[r] * OUTD + n], acc[r] + bias);
            }
        }
        __syncthreads();   // before next iteration rewrites per-wave tiles
    }
}

// ------------------------------- launcher ----------------------------------

extern "C" void kernel_launch(void* const* d_in, const int* in_sizes, int n_in,
                              void* d_out, int out_size, void* d_ws, size_t ws_size,
                              hipStream_t stream) {
    (void)in_sizes; (void)n_in; (void)out_size; (void)ws_size;

    const float*     x   = (const float*)d_in[0];
    const long long* ei  = (const long long*)d_in[1];
    const float*     ea  = (const float*)d_in[2];
    const float*     W1a = (const float*)d_in[3];
    const float*     b1a = (const float*)d_in[4];
    const float*     W1b = (const float*)d_in[5];
    const float*     b1b = (const float*)d_in[6];
    const float*     W2a = (const float*)d_in[7];
    const float*     b2a = (const float*)d_in[8];
    const float*     W2b = (const float*)d_in[9];
    const float*     b2b = (const float*)d_in[10];
    const float*     W3a = (const float*)d_in[11];
    const float*     b3a = (const float*)d_in[12];
    const float*     W3b = (const float*)d_in[13];
    const float*     b3b = (const float*)d_in[14];
    float* out = (float*)d_out;

    const int N = GNN_N_NODES;
    const int E = GNN_N_EDGES;

    // workspace layout: cnt f32[N] | accum f32[N*64] | feat16 f16[N*64] | ea16 f16[E*3]
    float*     cnt    = (float*)d_ws;
    float*     accum  = cnt + N;
    _Float16*  feat16 = (_Float16*)(accum + (size_t)N * 64);
    _Float16*  ea16   = feat16 + (size_t)N * 64;

    // zero cnt + accum (contiguous)
    {
        int total = N * 65;
        gnn_zero_kernel<<<(total + 255) / 256, 256, 0, stream>>>(cnt, total);
    }
    // edge_attr -> f16 once
    gnn_cvt_f16_kernel<<<(E * 3 + 255) / 256, 256, 0, stream>>>(ea, ea16, E * 3);
    // in-degree counts
    gnn_degree_kernel<<<(E + 255) / 256, 256, 0, stream>>>(ei, cnt, E);

    const int nTiles   = (E + 15) / 16;
    const int eblocks  = (nTiles + WPB * TILES_PER_WAVE - 1) / (WPB * TILES_PER_WAVE);
    const int ethreads = 32 * WPB;

    // layer 1: K=9 -> 64 -> 64  (f32 node input)
    gnn_edge_mlp_kernel<6, 64, 64><<<eblocks, ethreads, 0, stream>>>(
        x, ei, ea16, W1a, b1a, W1b, b1b, accum, E, nTiles);
    gnn_node_finish_kernel<64><<<(N * 64 + 255) / 256, 256, 0, stream>>>(
        cnt, accum, feat16, N * 64);

    // layer 2: K=67 -> 64 -> 64  (f16 node input)
    gnn_edge_mlp_kernel<64, 64, 64><<<eblocks, ethreads, 0, stream>>>(
        feat16, ei, ea16, W2a, b2a, W2b, b2b, accum, E, nTiles);
    gnn_node_finish_kernel<64><<<(N * 64 + 255) / 256, 256, 0, stream>>>(
        cnt, accum, feat16, N * 64);

    // layer 3: K=67 -> 32 -> 32  (accum reused with stride 32; region is zeroed)
    gnn_edge_mlp_kernel<64, 32, 32><<<eblocks, ethreads, 0, stream>>>(
        feat16, ei, ea16, W3a, b3a, W3b, b3b, accum, E, nTiles);

    // final mean (no relu) -> d_out
    gnn_final_kernel<32><<<(N * 32 + 255) / 256, 256, 0, stream>>>(
        cnt, accum, out, N * 32);
}
